// DecoderBlock_19284403159198
// MI455X (gfx1250) — matrix operand order repaired and checked by reading, not compiled, not run on hardware
//
#include <hip/hip_runtime.h>
#include <hip/hip_bf16.h>
#include <math.h>

// ---------------------------------------------------------------------------
// CDNA5 / gfx1250 decoder block:
//   - f32->f16 pre-pass (activations row-major, weights transposed [N][K])
//   - double-buffered WMMA GEMM, TDM (tensor_load_to_lds) staging when available
//   - flash attention on f16 operands, online softmax, WMMA for QK^T and PV
// ---------------------------------------------------------------------------

typedef __attribute__((ext_vector_type(16))) _Float16 v16h;
typedef __attribute__((ext_vector_type(8)))  _Float16 v8h;
typedef __attribute__((ext_vector_type(4)))  _Float16 v4h;
typedef __attribute__((ext_vector_type(8)))  float    v8f;
typedef unsigned int u32x4 __attribute__((ext_vector_type(4)));
typedef int          i32x8 __attribute__((ext_vector_type(8)));
typedef int          i32x4 __attribute__((ext_vector_type(4)));

#if defined(__gfx1250__) && __has_builtin(__builtin_amdgcn_tensor_load_to_lds)
#define USE_TDM 1
#else
#define USE_TDM 0
#endif

__device__ __forceinline__ v16h cat16(v8h a, v8h b) {
    return __builtin_shufflevector(a, b, 0,1,2,3,4,5,6,7,8,9,10,11,12,13,14,15);
}
__device__ __forceinline__ v8f wmma_f16(v16h a, v16h b, v8f c) {
    return __builtin_amdgcn_wmma_f32_16x16x32_f16(false, a, false, b,
                                                  (short)0, c, false, false);
}

// ---------------------------------------------------------------------------
// Pre-pass: f32 -> f16 row-major copy
// ---------------------------------------------------------------------------
__global__ __launch_bounds__(256)
void cvt_f32_f16(const float* __restrict__ X, _Float16* __restrict__ Y)
{
    const size_t i = ((size_t)blockIdx.x * 256 + threadIdx.x) * 4;
    const float4 v = *(const float4*)(X + i);
    v4h h = { (_Float16)v.x, (_Float16)v.y, (_Float16)v.z, (_Float16)v.w };
    *(v4h*)(Y + i) = h;
}

// Pre-pass: W[K][N] f32 -> Wt[N][K] f16 (tiled transpose)
__global__ __launch_bounds__(256)
void transpose_w_f16(const float* __restrict__ W, _Float16* __restrict__ Wt,
                     int K, int N)
{
    __shared__ _Float16 tile[32][33];
    const int n0 = blockIdx.x * 32, k0 = blockIdx.y * 32;
    const int tx = threadIdx.x & 31, ty = threadIdx.x >> 5;   // 32 x 8
    #pragma unroll
    for (int i = 0; i < 32; i += 8)
        tile[ty + i][tx] = (_Float16)W[(size_t)(k0 + ty + i) * N + n0 + tx];
    __syncthreads();
    #pragma unroll
    for (int i = 0; i < 32; i += 8)
        Wt[(size_t)(n0 + ty + i) * K + k0 + tx] = tile[tx][ty + i];
}

// ---------------------------------------------------------------------------
// TDM tile descriptor: 2D tile [128 rows][32 halves], row stride K halves,
// LDS padding 4 DWORDs every 16 DWORDs -> 40-half LDS row stride.
// 6-arg builtin form (clang-23 / therock-10.0): (g0,g1,g2,g3,g4,cpol)
// ---------------------------------------------------------------------------
#if USE_TDM
__device__ __forceinline__ void tdm_load_tile(const _Float16* gsrc, void* lds_dst, int K)
{
    const unsigned long long ga = (unsigned long long)(size_t)gsrc;
    const unsigned lds = (unsigned)(size_t)lds_dst;
    u32x4 g0;
    g0[0] = 1u;                                           // count=1 (valid D#)
    g0[1] = lds;                                          // lds_addr
    g0[2] = (unsigned)(ga & 0xFFFFFFFFu);                 // global_addr[31:0]
    g0[3] = (unsigned)((ga >> 32) & 0x01FFFFFFu) | (2u << 30);  // [56:32] | type=2
    i32x8 g1;
    g1[0] = (int)((1u << 16) | (1u << 20) | (3u << 22) | (3u << 25));
            // data_size=2B, pad_enable, pad_interval=16dw, pad_amount=4dw
    g1[1] = (int)(((unsigned)K & 0xFFFFu) << 16);         // tensor_dim0 lo16
    g1[2] = (int)((((unsigned)K >> 16) & 0xFFFFu) |
                  ((1u << 20 & 0xFFFFu) << 16));          // dim0 hi | dim1 lo16
    g1[3] = (int)(((1u << 20) >> 16) | (32u << 16));      // dim1 hi | tile_dim0=32
    g1[4] = (int)(128u);                                  // tile_dim1=128, tile_dim2=0
    g1[5] = (int)(unsigned)K;                             // tensor_dim0_stride lo32
    g1[6] = 0;
    g1[7] = 0;
    const i32x4 z4 = {0, 0, 0, 0};
    const i32x8 z8 = {0, 0, 0, 0, 0, 0, 0, 0};
    __builtin_amdgcn_tensor_load_to_lds(g0, g1, z4, z4, z8, 0);
}
#endif

// ---------------------------------------------------------------------------
// GEMM: out = act( Ah[M,K](f16) @ Wt[N,K]^T(f16) + bias (+ resid f32) )
// Tile 128x128, K-step 32, 8 waves, double-buffered LDS, TDM staging.
// Optional f32 and/or f16 outputs.
// ---------------------------------------------------------------------------
#define GBM 128
#define GBN 128
#define GBK 32
#define LDT 40   // 32 + 8 pad halves (80 B row stride)

__global__ __launch_bounds__(256)
void gemm_wmma_f16(const _Float16* __restrict__ Ah, const _Float16* __restrict__ Wt,
                   const float* __restrict__ bias, const float* __restrict__ resid,
                   float* __restrict__ Cf, _Float16* __restrict__ Ch,
                   int M, int K, int Nc, int do_gelu)
{
    __shared__ __align__(16) _Float16 As[2][GBM * LDT];
    __shared__ __align__(16) _Float16 Bs[2][GBN * LDT];

    const int t    = threadIdx.x;
    const int lane = t & 31;
    const int w    = t >> 5;
    const int l15  = lane & 15;
    const int hi   = lane >> 4;
    const int m0   = blockIdx.y * GBM;
    const int n0   = blockIdx.x * GBN;
    const int wm   = (w >> 1) * 32;
    const int wn   = (w & 1) * 64;

    const _Float16* Ag = Ah + (size_t)m0 * K;   // tile row origin
    const _Float16* Bg = Wt + (size_t)n0 * K;

    const int srow = t >> 2;            // staging row (2 chunks/thread/tile)
    const int sc8  = (t & 3) * 8;       // halves within row

    // ---- prologue: stage k0 = 0 into buffer 0 ----
#if USE_TDM
    if (w == 0) {
        tdm_load_tile(Ag, &As[0][0], K);
        tdm_load_tile(Bg, &Bs[0][0], K);
        __builtin_amdgcn_s_wait_tensorcnt(0);
    }
#else
    #pragma unroll
    for (int it = 0; it < 2; ++it) {
        const int r = srow + it * 64;
        *(v8h*)(&As[0][r * LDT + sc8]) = *(const v8h*)(Ag + (size_t)r * K + sc8);
        *(v8h*)(&Bs[0][r * LDT + sc8]) = *(const v8h*)(Bg + (size_t)r * K + sc8);
    }
#endif
    __syncthreads();

    v8f acc[2][4] = {};
    const int kb = hi * 8;

    for (int k0 = 0; k0 < K; k0 += GBK) {
        const int cur = (k0 >> 5) & 1;
        const int nxt = cur ^ 1;
        const bool has_next = (k0 + GBK) < K;

#if USE_TDM
        if (has_next && w == 0) {       // fire-and-forget next tile
            tdm_load_tile(Ag + k0 + GBK, &As[nxt][0], K);
            tdm_load_tile(Bg + k0 + GBK, &Bs[nxt][0], K);
        }
#else
        v8h pa[2], pb[2];
        if (has_next) {
            #pragma unroll
            for (int it = 0; it < 2; ++it) {
                const int r = srow + it * 64;
                pa[it] = *(const v8h*)(Ag + (size_t)r * K + k0 + GBK + sc8);
                pb[it] = *(const v8h*)(Bg + (size_t)r * K + k0 + GBK + sc8);
            }
        }
#endif
        // ---- compute on current buffer ----
        v16h af[2], bf[4];
        #pragma unroll
        for (int i = 0; i < 2; ++i) {
            const _Float16* p = &As[cur][(wm + i * 16 + l15) * LDT];
            af[i] = cat16(*(const v8h*)(p + kb), *(const v8h*)(p + kb + 16));
        }
        #pragma unroll
        for (int j = 0; j < 4; ++j) {
            const _Float16* p = &Bs[cur][(wn + j * 16 + l15) * LDT + hi * 8];
            bf[j] = cat16(*(const v8h*)p, *(const v8h*)(p + 16));
        }
        #pragma unroll
        for (int i = 0; i < 2; ++i)
            #pragma unroll
            for (int j = 0; j < 4; ++j)
                acc[i][j] = wmma_f16(af[i], bf[j], acc[i][j]);

        // ---- finish staging of next buffer ----
#if USE_TDM
        if (has_next && w == 0) __builtin_amdgcn_s_wait_tensorcnt(0);
#else
        if (has_next) {
            #pragma unroll
            for (int it = 0; it < 2; ++it) {
                const int r = srow + it * 64;
                *(v8h*)(&As[nxt][r * LDT + sc8]) = pa[it];
                *(v8h*)(&Bs[nxt][r * LDT + sc8]) = pb[it];
            }
        }
#endif
        __syncthreads();
    }

    // ---- epilogue ----
    #pragma unroll
    for (int i = 0; i < 2; ++i) {
        #pragma unroll
        for (int j = 0; j < 4; ++j) {
            const int col = n0 + wn + j * 16 + l15;
            const float bv = bias[col];
            #pragma unroll
            for (int r = 0; r < 8; ++r) {
                const int row = m0 + wm + i * 16 + hi * 8 + r;
                float v = acc[i][j][r] + bv;
                if (resid) v += resid[(size_t)row * Nc + col];
                if (do_gelu) v = 0.5f * v * (1.0f + erff(v * 0.70710678118654752440f));
                if (Cf) Cf[(size_t)row * Nc + col] = v;
                if (Ch) Ch[(size_t)row * Nc + col] = (_Float16)v;
            }
        }
    }
}

// ---------------------------------------------------------------------------
// Flash attention on f16 operands. Grid (seqQ/128, H, B), 256 threads.
// ---------------------------------------------------------------------------
#define AKT 64
#define KSTR 72
#define VSTR 72

__global__ __launch_bounds__(256)
void attn_flash_wmma(const _Float16* __restrict__ Qp, const _Float16* __restrict__ Kp,
                     const _Float16* __restrict__ Vp, _Float16* __restrict__ Oh,
                     int seqQ, int seqK, int sQ, int sKV, int sO,
                     int causal, float scale)
{
    __shared__ __align__(16) _Float16 Ks[AKT * KSTR];      // [key][d]
    __shared__ __align__(16) _Float16 Vs[64 * VSTR];       // [d][key]
    __shared__ __align__(16) _Float16 Ps[8 * 16 * AKT];    // per-wave P slab

    const int t    = threadIdx.x;
    const int lane = t & 31;
    const int w    = t >> 5;
    const int l15  = lane & 15;
    const int hi   = lane >> 4;
    const int b    = blockIdx.z;
    const int h    = blockIdx.y;
    const int q0   = blockIdx.x * 128;

    // --- Q fragments (registers): contiguous 8-half runs -> 2x v8h each ---
    const int qrow = q0 + w * 16 + l15;
    const _Float16* Qr = Qp + (size_t)(b * seqQ + qrow) * sQ + h * 64;
    v16h qf[2];
    #pragma unroll
    for (int c = 0; c < 2; ++c) {
        const _Float16* p = Qr + c * 32 + hi * 8;
        qf[c] = cat16(*(const v8h*)p, *(const v8h*)(p + 16));
    }

    v8f  o[4] = {};
    float m[8], l[8];
    #pragma unroll
    for (int r = 0; r < 8; ++r) { m[r] = -1e30f; l[r] = 0.f; }

    int kend = seqK;
    if (causal) { int lim = q0 + 128; kend = lim < seqK ? lim : seqK; }

    const int skey = t >> 3;          // staging: 32 keys x 8 chunks
    const int sc8  = (t & 7) * 8;

    for (int kb0 = 0; kb0 < kend; kb0 += AKT) {
        // ---- stage K [key][d] (copy) and V transposed [d][key] ----
        {
            const size_t rb = (size_t)(b * seqK + kb0 + skey) * sKV + h * 64;
            const v8h kk = *(const v8h*)(Kp + rb + sc8);
            *(v8h*)(&Ks[skey * KSTR + sc8]) = kk;
            const v8h vv = *(const v8h*)(Vp + rb + sc8);
            #pragma unroll
            for (int j = 0; j < 8; ++j)
                Vs[(sc8 + j) * VSTR + skey] = vv[j];
        }
        __syncthreads();

        // ---- S = Q K^T ----
        v8f s[4];
        #pragma unroll
        for (int j = 0; j < 4; ++j) {
            v8f z = {};
            #pragma unroll
            for (int c = 0; c < 2; ++c) {
                const _Float16* p = Ks + (j * 16 + l15) * KSTR + c * 32 + hi * 8;
                z = wmma_f16(qf[c], cat16(*(const v8h*)p, *(const v8h*)(p + 16)), z);
            }
            s[j] = z;
        }

        // ---- online softmax (rows live in 16-lane halves) ----
        float alpha[8], mnew[8], psum[8];
        #pragma unroll
        for (int r = 0; r < 8; ++r) {
            const int qg = q0 + w * 16 + hi * 8 + r;
            float mx = -1e30f;
            #pragma unroll
            for (int j = 0; j < 4; ++j) {
                float sv = s[j][r] * scale;
                const int kg = kb0 + j * 16 + l15;
                if (causal && kg > qg) sv = -1e30f;
                s[j][r] = sv;
                mx = fmaxf(mx, sv);
            }
            #pragma unroll
            for (int off = 8; off; off >>= 1) mx = fmaxf(mx, __shfl_xor(mx, off, 32));
            mnew[r]  = fmaxf(m[r], mx);
            alpha[r] = expf(m[r] - mnew[r]);
            float ps = 0.f;
            #pragma unroll
            for (int j = 0; j < 4; ++j) {
                const float sv = s[j][r];
                const float p  = (sv <= -1e29f) ? 0.f : expf(sv - mnew[r]);
                s[j][r] = p;
                ps += p;
            }
            #pragma unroll
            for (int off = 8; off; off >>= 1) ps += __shfl_xor(ps, off, 32);
            psum[r] = ps;
        }

        // ---- spill P to wave-private LDS (DS in-order within wave) ----
        _Float16* Pw = Ps + w * (16 * AKT);
        #pragma unroll
        for (int j = 0; j < 4; ++j)
            #pragma unroll
            for (int r = 0; r < 8; ++r)
                Pw[(hi * 8 + r) * AKT + j * 16 + l15] = (_Float16)s[j][r];

        #pragma unroll
        for (int r = 0; r < 8; ++r) {
            #pragma unroll
            for (int n = 0; n < 4; ++n) o[n][r] *= alpha[r];
            l[r] = l[r] * alpha[r] + psum[r];
            m[r] = mnew[r];
        }

        // ---- O += P @ V ----
        #pragma unroll
        for (int ks = 0; ks < 2; ++ks) {
            const _Float16* pp = Pw + l15 * AKT + ks * 32 + hi * 8;
            v16h pf = cat16(*(const v8h*)pp, *(const v8h*)(pp + 16));
            #pragma unroll
            for (int n = 0; n < 4; ++n) {
                const _Float16* vp = Vs + (n * 16 + l15) * VSTR + ks * 32 + hi * 8;
                o[n] = wmma_f16(pf, cat16(*(const v8h*)vp, *(const v8h*)(vp + 16)), o[n]);
            }
        }
        __syncthreads();
    }

    // ---- normalize & store (f16) ----
    #pragma unroll
    for (int n = 0; n < 4; ++n)
        #pragma unroll
        for (int r = 0; r < 8; ++r) {
            const int row = q0 + w * 16 + hi * 8 + r;
            const float inv = 1.0f / l[r];
            Oh[(size_t)(b * seqQ + row) * sO + h * 64 + n * 16 + l15] =
                (_Float16)(o[n][r] * inv);
        }
}

// ---------------------------------------------------------------------------
// Row LayerNorm: f32 in, f32 out (+ optional f16 copy).
// ---------------------------------------------------------------------------
__global__ __launch_bounds__(256)
void layernorm_row(const float* __restrict__ X, const float* __restrict__ g,
                   const float* __restrict__ bt, float* __restrict__ Y,
                   _Float16* __restrict__ Yh, int Cn)
{
    __shared__ float red[16];
    const int row = blockIdx.x;
    const float* x = X + (size_t)row * Cn;
    const int t = threadIdx.x, w = t >> 5, lane = t & 31;

    float lv[4], s = 0.f, s2 = 0.f;
    #pragma unroll
    for (int i = 0; i < 4; ++i) {
        const float v = x[t + i * 256];
        lv[i] = v; s += v; s2 += v * v;
    }
    #pragma unroll
    for (int off = 16; off; off >>= 1) {
        s  += __shfl_xor(s,  off, 32);
        s2 += __shfl_xor(s2, off, 32);
    }
    if (lane == 0) { red[w] = s; red[8 + w] = s2; }
    __syncthreads();
    if (w == 0) {
        float a  = (lane < 8) ? red[lane]     : 0.f;
        float a2 = (lane < 8) ? red[8 + lane] : 0.f;
        #pragma unroll
        for (int off = 4; off; off >>= 1) {
            a  += __shfl_xor(a,  off, 32);
            a2 += __shfl_xor(a2, off, 32);
        }
        if (lane == 0) { red[0] = a; red[8] = a2; }
    }
    __syncthreads();
    const float mean = red[0] / Cn;
    const float var  = red[8] / Cn - mean * mean;
    const float rstd = rsqrtf(var + 1e-5f);
    #pragma unroll
    for (int i = 0; i < 4; ++i) {
        const int c = t + i * 256;
        const float v = (lv[i] - mean) * rstd * g[c] + bt[c];
        Y[(size_t)row * Cn + c] = v;
        if (Yh) Yh[(size_t)row * Cn + c] = (_Float16)v;
    }
}

// ---------------------------------------------------------------------------
// Host orchestration
// ---------------------------------------------------------------------------
extern "C" void kernel_launch(void* const* d_in, const int* in_sizes, int n_in,
                              void* d_out, int out_size, void* d_ws, size_t ws_size,
                              hipStream_t stream)
{
    (void)in_sizes; (void)n_in; (void)out_size; (void)ws_size;
    constexpr int Bb = 8, Nn = 1024, Tt = 512, Cc = 1024, Hh = 16;
    constexpr int M1 = Bb * Nn;   // 8192
    constexpr int M2 = Bb * Tt;   // 4096
    const float scale = 0.125f;

    const float* prev   = (const float*)d_in[0];
    const float* enc    = (const float*)d_in[1];
    const float* sa_w   = (const float*)d_in[2];
    const float* sa_b   = (const float*)d_in[3];
    const float* sa_pw  = (const float*)d_in[4];
    const float* sa_pb  = (const float*)d_in[5];
    const float* caq_w  = (const float*)d_in[6];
    const float* caq_b  = (const float*)d_in[7];
    const float* cakv_w = (const float*)d_in[8];
    const float* cakv_b = (const float*)d_in[9];
    const float* ca_pw  = (const float*)d_in[10];
    const float* ca_pb  = (const float*)d_in[11];
    const float* fc_w   = (const float*)d_in[12];
    const float* fc_b   = (const float*)d_in[13];
    const float* proj_w = (const float*)d_in[14];
    const float* proj_b = (const float*)d_in[15];
    const float* ln1_g  = (const float*)d_in[16];
    const float* ln1_b  = (const float*)d_in[17];
    const float* ln2_g  = (const float*)d_in[18];
    const float* ln2_b  = (const float*)d_in[19];
    const float* ln3_g  = (const float*)d_in[20];
    const float* ln3_b  = (const float*)d_in[21];

    // ---- workspace layout ----
    char* wsb = (char*)d_ws;
    size_t off = 0;
    auto alloc_h = [&](size_t n) { _Float16* p = (_Float16*)(wsb + off);
                                   off += ((n * 2 + 255) & ~(size_t)255); return p; };
    auto alloc_f = [&](size_t n) { float* p = (float*)(wsb + off);
                                   off += ((n * 4 + 255) & ~(size_t)255); return p; };

    _Float16* prev_h  = alloc_h((size_t)M1 * Cc);
    _Float16* enc_h   = alloc_h((size_t)M2 * Cc);
    _Float16* sa_wt   = alloc_h((size_t)3 * Cc * Cc);
    _Float16* sa_pwt  = alloc_h((size_t)Cc * Cc);
    _Float16* caq_wt  = alloc_h((size_t)Cc * Cc);
    _Float16* cakv_wt = alloc_h((size_t)2 * Cc * Cc);
    _Float16* ca_pwt  = alloc_h((size_t)Cc * Cc);
    _Float16* fc_wt   = alloc_h((size_t)4 * Cc * Cc);
    _Float16* proj_wt = alloc_h((size_t)4 * Cc * Cc);
    _Float16* qkv_h   = alloc_h((size_t)M1 * 3 * Cc);   // reused: q2 | kv | y2
    _Float16* y1_h    = alloc_h((size_t)M1 * Cc);       // reused: x2_h
    float*    x1      = alloc_f((size_t)M1 * Cc);       // reused: x3
    _Float16* x1_h    = alloc_h((size_t)M1 * Cc);
    float*    x2      = alloc_f((size_t)M1 * Cc);
    _Float16* h_h     = alloc_h((size_t)M1 * 4 * Cc);

    _Float16* q2_h = qkv_h;
    _Float16* kv_h = qkv_h + (size_t)M1 * Cc;
    _Float16* y2_h = qkv_h + (size_t)2 * M1 * Cc;
    _Float16* x2_h = y1_h;
    float*    x3   = x1;

    dim3 blk(256);
    auto gg = [](int M, int Nc) { return dim3(Nc / GBN, M / GBM, 1); };
    auto gt = [](int K, int N)  { return dim3(N / 32, K / 32, 1); };

    // ---- pre-pass conversions ----
    cvt_f32_f16<<<(M1 * Cc) / 1024, blk, 0, stream>>>(prev, prev_h);
    cvt_f32_f16<<<(M2 * Cc) / 1024, blk, 0, stream>>>(enc, enc_h);
    transpose_w_f16<<<gt(Cc, 3 * Cc), blk, 0, stream>>>(sa_w,   sa_wt,   Cc, 3 * Cc);
    transpose_w_f16<<<gt(Cc, Cc),     blk, 0, stream>>>(sa_pw,  sa_pwt,  Cc, Cc);
    transpose_w_f16<<<gt(Cc, Cc),     blk, 0, stream>>>(caq_w,  caq_wt,  Cc, Cc);
    transpose_w_f16<<<gt(Cc, 2 * Cc), blk, 0, stream>>>(cakv_w, cakv_wt, Cc, 2 * Cc);
    transpose_w_f16<<<gt(Cc, Cc),     blk, 0, stream>>>(ca_pw,  ca_pwt,  Cc, Cc);
    transpose_w_f16<<<gt(Cc, 4 * Cc), blk, 0, stream>>>(fc_w,   fc_wt,   Cc, 4 * Cc);
    transpose_w_f16<<<gt(4 * Cc, Cc), blk, 0, stream>>>(proj_w, proj_wt, 4 * Cc, Cc);

    // 1) qkv = prev @ sa_w + sa_b          (f16 out only)
    gemm_wmma_f16<<<gg(M1, 3 * Cc), blk, 0, stream>>>(prev_h, sa_wt, sa_b, nullptr,
                                                      nullptr, qkv_h, M1, Cc, 3 * Cc, 0);
    // 2) causal self-attention -> y1 (f16)
    attn_flash_wmma<<<dim3(Nn / 128, Hh, Bb), blk, 0, stream>>>(
        qkv_h, qkv_h + Cc, qkv_h + 2 * Cc, y1_h, Nn, Nn, 3 * Cc, 3 * Cc, Cc, 1, scale);
    // 3) x1 = y1 @ sa_pw + sa_pb  (f32) ; 4) LN1 -> x1 (f32) + x1_h (f16)
    gemm_wmma_f16<<<gg(M1, Cc), blk, 0, stream>>>(y1_h, sa_pwt, sa_pb, nullptr,
                                                  x1, nullptr, M1, Cc, Cc, 0);
    layernorm_row<<<M1, blk, 0, stream>>>(x1, ln1_g, ln1_b, x1, x1_h, Cc);
    // 5) q2 = x1ln @ caq_w + caq_b (f16)
    gemm_wmma_f16<<<gg(M1, Cc), blk, 0, stream>>>(x1_h, caq_wt, caq_b, nullptr,
                                                  nullptr, q2_h, M1, Cc, Cc, 0);
    // 6) kv = enc @ cakv_w + cakv_b (f16)
    gemm_wmma_f16<<<gg(M2, 2 * Cc), blk, 0, stream>>>(enc_h, cakv_wt, cakv_b, nullptr,
                                                      nullptr, kv_h, M2, Cc, 2 * Cc, 0);
    // 7) cross-attention -> y2 (f16)
    attn_flash_wmma<<<dim3(Nn / 128, Hh, Bb), blk, 0, stream>>>(
        q2_h, kv_h, kv_h + Cc, y2_h, Nn, Tt, Cc, 2 * Cc, Cc, 0, scale);
    // 8) x2 = x1ln + y2 @ ca_pw + ca_pb (f32) ; 9) LN2 -> x2 + x2_h
    gemm_wmma_f16<<<gg(M1, Cc), blk, 0, stream>>>(y2_h, ca_pwt, ca_pb, x1,
                                                  x2, nullptr, M1, Cc, Cc, 0);
    layernorm_row<<<M1, blk, 0, stream>>>(x2, ln2_g, ln2_b, x2, x2_h, Cc);
    // 10) h = gelu(x2ln @ fc_w + fc_b) (f16)
    gemm_wmma_f16<<<gg(M1, 4 * Cc), blk, 0, stream>>>(x2_h, fc_wt, fc_b, nullptr,
                                                      nullptr, h_h, M1, Cc, 4 * Cc, 1);
    // 11) x3 = x2ln + h @ proj_w + proj_b (f32)
    gemm_wmma_f16<<<gg(M1, Cc), blk, 0, stream>>>(h_h, proj_wt, proj_b, x2,
                                                  x3, nullptr, M1, 4 * Cc, Cc, 0);
    // 12) out = LN3(x3)
    layernorm_row<<<M1, blk, 0, stream>>>(x3, ln3_g, ln3_b, (float*)d_out, nullptr, Cc);
}
